// WindowMultiHeadAttention_56143812494097
// MI455X (gfx1250) — matrix operand order repaired and checked by reading, not compiled, not run on hardware
//
#include <hip/hip_runtime.h>
#include <hip/hip_bf16.h>

typedef __attribute__((ext_vector_type(16))) __bf16 v16bf;
typedef __attribute__((ext_vector_type(8)))  float  v8f;

union FragBF { v16bf v; uint4 u[2]; };

typedef unsigned short ush;

__device__ __forceinline__ unsigned int f2bf_u(float f) {
    unsigned int u = __float_as_uint(f);
    u += 0x7FFFu + ((u >> 16) & 1u);
    return u >> 16;
}
__device__ __forceinline__ ush f2bf(float f) { return (ush)f2bf_u(f); }
__device__ __forceinline__ unsigned int pack2bf(float lo, float hi) {
    return f2bf_u(lo) | (f2bf_u(hi) << 16);
}

// 16x32 bf16 A-fragment (row-major source, ld in elements).
// lane L (0-15): row L, K = 0..7 (elems 0-7) and 16..23 (elems 8-15)
// lane L (16-31): row L-16, K = 8..15 and 24..31
__device__ __forceinline__ v16bf loadA32(const ush* base, int ld) {
    int lane = threadIdx.x & 31;
    int row  = lane & 15, half = lane >> 4;
    const ush* p = base + row * ld + half * 8;
    FragBF f;
    f.u[0] = *(const uint4*)(p);
    f.u[1] = *(const uint4*)(p + 16);
    return f.v;
}

// 32x16 bf16 B-fragment loaded from B^T storage [N][K] row-major (K contiguous).
// lane L: column n = L&15; K = (L>>4)*8 .. +7 (elems 0-7) and 16+(L>>4)*8 .. +7 (elems 8-15)
__device__ __forceinline__ v16bf loadB32(const ush* baseT, int ld) {
    int lane = threadIdx.x & 31;
    int col  = lane & 15, half = lane >> 4;
    const ush* p = baseT + col * ld + half * 8;
    FragBF f;
    f.u[0] = *(const uint4*)(p);
    f.u[1] = *(const uint4*)(p + 16);
    return f.v;
}

#define WMMA_BF16(A, B, C) \
    __builtin_amdgcn_wmma_f32_16x16x32_bf16(false, (A), false, (B), (short)0, (C), false, false)

// ---------------- conversion kernels (8 elems / thread) ----------------

// window partition + bf16 convert: x[B,4096,512] f32 -> x_win[65536,512] bf16
__global__ void __launch_bounds__(256) win_part_bf16(const float* __restrict__ x,
                                                     ush* __restrict__ xw) {
    size_t idx = ((size_t)blockIdx.x * 256 + threadIdx.x) * 8;   // < 65536*512
    int c   = (int)(idx & 511);
    int m   = (int)(idx >> 9);
    int win = m >> 6, tok = m & 63;
    int b   = win >> 6, wi = win & 63;
    int wy = wi >> 3, wx = wi & 7, ty = tok >> 3, tx = tok & 7;
    size_t src = ((size_t)b * 4096 + (size_t)((wy * 8 + ty) * 64 + wx * 8 + tx)) * 512 + c;
    const float4* p = (const float4*)(x + src);
    float4 f0 = p[0], f1 = p[1];
    uint4 o;
    o.x = pack2bf(f0.x, f0.y);
    o.y = pack2bf(f0.z, f0.w);
    o.z = pack2bf(f1.x, f1.y);
    o.w = pack2bf(f1.z, f1.w);
    *(uint4*)(xw + idx) = o;
}

__global__ void __launch_bounds__(256) cvt_bf16(const float* __restrict__ in,
                                                ush* __restrict__ out, int n8) {
    int i = blockIdx.x * 256 + threadIdx.x;      // i indexes groups of 8
    if (i >= n8) return;
    const float4* p = (const float4*)(in + (size_t)i * 8);
    float4 f0 = p[0], f1 = p[1];
    uint4 o;
    o.x = pack2bf(f0.x, f0.y);
    o.y = pack2bf(f0.z, f0.w);
    o.z = pack2bf(f1.x, f1.y);
    o.w = pack2bf(f1.z, f1.w);
    *(uint4*)(out + (size_t)i * 8) = o;
}

// ---------------- QKV GEMM: 64M x 64N per wave ----------------
// A: x_win bf16 [65536,512]; W: qkv_w bf16 [1536,512] (== B^T layout)
__global__ void __launch_bounds__(256) qkv_gemm(const ush* __restrict__ A,
                                                const ush* __restrict__ W,
                                                const float* __restrict__ bias,
                                                ush* __restrict__ qb,
                                                ush* __restrict__ kb,
                                                ush* __restrict__ vTb) {
    const int NG = 24;                                   // 1536 / 64
    int wave  = blockIdx.x * 8 + (threadIdx.x >> 5);
    int mt64  = wave / NG;                               // 0..1023
    int ng    = wave - mt64 * NG;
    int lane  = threadIdx.x & 31;
    int half  = lane >> 4, nl = lane & 15;

    const ush* Ab = A + (size_t)mt64 * 64 * 512;
    const ush* Wb = W + (size_t)ng * 64 * 512;

    v8f acc[4][4];
    #pragma unroll
    for (int mi = 0; mi < 4; ++mi)
        #pragma unroll
        for (int ni = 0; ni < 4; ++ni) acc[mi][ni] = (v8f){};

    for (int kk = 0; kk < 512; kk += 32) {
        // prefetch A one K-step ahead (64 rows via 2 lane-strided prefetches)
        if (kk + 32 < 512) {
            __builtin_prefetch(Ab + (size_t)lane * 512 + kk + 32, 0, 1);
            __builtin_prefetch(Ab + (size_t)(lane + 32) * 512 + kk + 32, 0, 1);
        }
        v16bf a0 = loadA32(Ab + kk,            512);
        v16bf a1 = loadA32(Ab + 16 * 512 + kk, 512);
        v16bf a2 = loadA32(Ab + 32 * 512 + kk, 512);
        v16bf a3 = loadA32(Ab + 48 * 512 + kk, 512);
        v16bf b0 = loadB32(Wb + kk,            512);
        v16bf b1 = loadB32(Wb + 16 * 512 + kk, 512);
        v16bf b2 = loadB32(Wb + 32 * 512 + kk, 512);
        v16bf b3 = loadB32(Wb + 48 * 512 + kk, 512);
        acc[0][0] = WMMA_BF16(a0, b0, acc[0][0]);
        acc[1][0] = WMMA_BF16(a1, b0, acc[1][0]);
        acc[2][0] = WMMA_BF16(a2, b0, acc[2][0]);
        acc[3][0] = WMMA_BF16(a3, b0, acc[3][0]);
        acc[0][1] = WMMA_BF16(a0, b1, acc[0][1]);
        acc[1][1] = WMMA_BF16(a1, b1, acc[1][1]);
        acc[2][1] = WMMA_BF16(a2, b1, acc[2][1]);
        acc[3][1] = WMMA_BF16(a3, b1, acc[3][1]);
        acc[0][2] = WMMA_BF16(a0, b2, acc[0][2]);
        acc[1][2] = WMMA_BF16(a1, b2, acc[1][2]);
        acc[2][2] = WMMA_BF16(a2, b2, acc[2][2]);
        acc[3][2] = WMMA_BF16(a3, b2, acc[3][2]);
        acc[0][3] = WMMA_BF16(a0, b3, acc[0][3]);
        acc[1][3] = WMMA_BF16(a1, b3, acc[1][3]);
        acc[2][3] = WMMA_BF16(a2, b3, acc[2][3]);
        acc[3][3] = WMMA_BF16(a3, b3, acc[3][3]);
    }

    const float qscale = 0.17677669529663687f;           // 1/sqrt(32)
    #pragma unroll
    for (int nt = 0; nt < 4; ++nt) {
        int n0   = ng * 64 + nt * 16;
        int qkvi = n0 >> 9;                              // 0=q 1=k 2=v
        int nin  = n0 & 511;
        int head = nin >> 5;
        int hd   = (nin & 31) + nl;                      // head-dim index
        float bval = bias[n0 + nl];
        #pragma unroll
        for (int mi = 0; mi < 4; ++mi) {
            #pragma unroll
            for (int e = 0; e < 8; ++e) {
                int m = mt64 * 64 + mi * 16 + e + half * 8;
                int win = m >> 6, tok = m & 63;
                size_t wh = (size_t)(win * 16 + head);
                float val = acc[mi][nt][e] + bval;
                if (qkvi == 0)      qb [(wh * 64 + tok) * 32 + hd] = f2bf(val * qscale);
                else if (qkvi == 1) kb [(wh * 64 + tok) * 32 + hd] = f2bf(val);
                else                vTb[(wh * 32 + hd) * 64 + tok] = f2bf(val);
            }
        }
    }
}

// ---------------- attention: one wave per (window, head) ----------------
__global__ void __launch_bounds__(256) attn_kernel(const ush* __restrict__ q,
                                                   const ush* __restrict__ k,
                                                   const ush* __restrict__ vT,
                                                   const float* __restrict__ mask,
                                                   ush* __restrict__ ao) {
    __shared__ __align__(16) ush lds[8][16 * 64];        // per-wave prob tile (bf16)
    int wib  = threadIdx.x >> 5;
    int wh   = blockIdx.x * 8 + wib;                     // 0..16383
    int win  = wh >> 4, head = wh & 15;
    int lane = threadIdx.x & 31;
    int half = lane >> 4, nl = lane & 15;

    const ush* qbp = q  + (size_t)wh * (64 * 32);
    const ush* kbp = k  + (size_t)wh * (64 * 32);
    const ush* vbp = vT + (size_t)wh * (32 * 64);
    const float* mb = mask + (size_t)(win & 63) * 4096;  // [64,64]

    for (int mt = 0; mt < 4; ++mt) {
        // ---- scores = q @ k^T (q pre-scaled) + mask ----
        v16bf aq = loadA32(qbp + (mt * 16) * 32, 32);
        v8f s[4] = {};
        #pragma unroll
        for (int nt = 0; nt < 4; ++nt) {
            v16bf bk = loadB32(kbp + (nt * 16) * 32, 32);
            s[nt] = WMMA_BF16(aq, bk, s[nt]);
        }
        #pragma unroll
        for (int nt = 0; nt < 4; ++nt)
            #pragma unroll
            for (int e = 0; e < 8; ++e)
                s[nt][e] += mb[(mt * 16 + e + half * 8) * 64 + nt * 16 + nl];

        // ---- softmax over 64 keys (rows live in 16-lane halves) ----
        float rmax[8], rinv[8];
        #pragma unroll
        for (int e = 0; e < 8; ++e) {
            float mx = s[0][e];
            mx = fmaxf(mx, s[1][e]); mx = fmaxf(mx, s[2][e]); mx = fmaxf(mx, s[3][e]);
            mx = fmaxf(mx, __shfl_xor(mx, 1, 32));
            mx = fmaxf(mx, __shfl_xor(mx, 2, 32));
            mx = fmaxf(mx, __shfl_xor(mx, 4, 32));
            mx = fmaxf(mx, __shfl_xor(mx, 8, 32));
            rmax[e] = mx;
        }
        #pragma unroll
        for (int e = 0; e < 8; ++e) {
            float sm = 0.f;
            int r = e + half * 8;
            #pragma unroll
            for (int nt = 0; nt < 4; ++nt) {
                float p = __expf(s[nt][e] - rmax[e]);
                sm += p;
                lds[wib][r * 64 + nt * 16 + nl] = f2bf(p);   // unnormalized probs
            }
            sm += __shfl_xor(sm, 1, 32);
            sm += __shfl_xor(sm, 2, 32);
            sm += __shfl_xor(sm, 4, 32);
            sm += __shfl_xor(sm, 8, 32);
            rinv[e] = 1.0f / sm;
        }
        __syncthreads();

        // ---- out = P @ v  (K = 64 tokens = 2 WMMA steps) ----
        v8f o0 = {}, o1 = {};
        #pragma unroll
        for (int ks = 0; ks < 2; ++ks) {
            v16bf ap = loadA32(&lds[wib][ks * 32], 64);
            v16bf b0 = loadB32(vbp + ks * 32, 64);
            v16bf b1 = loadB32(vbp + 16 * 64 + ks * 32, 64);
            o0 = WMMA_BF16(ap, b0, o0);
            o1 = WMMA_BF16(ap, b1, o1);
        }
        __syncthreads();

        // ---- normalize + store bf16 into [65536, 512] ----
        #pragma unroll
        for (int e = 0; e < 8; ++e) {
            int tok = mt * 16 + e + half * 8;
            size_t row = (size_t)win * 64 + tok;
            ao[row * 512 + head * 32 + nl]      = f2bf(o0[e] * rinv[e]);
            ao[row * 512 + head * 32 + 16 + nl] = f2bf(o1[e] * rinv[e]);
        }
    }
}

// ---------------- proj GEMM (64M x 64N per wave) + window reverse ----------------
__global__ void __launch_bounds__(256) proj_gemm(const ush* __restrict__ A,
                                                 const ush* __restrict__ W,
                                                 const float* __restrict__ bias,
                                                 float* __restrict__ out) {
    const int NG = 8;                                    // 512 / 64
    int wave  = blockIdx.x * 8 + (threadIdx.x >> 5);
    int mt64  = wave / NG;
    int ng    = wave - mt64 * NG;
    int lane  = threadIdx.x & 31;
    int half  = lane >> 4, nl = lane & 15;

    const ush* Ab = A + (size_t)mt64 * 64 * 512;
    const ush* Wb = W + (size_t)ng * 64 * 512;

    v8f acc[4][4];
    #pragma unroll
    for (int mi = 0; mi < 4; ++mi)
        #pragma unroll
        for (int ni = 0; ni < 4; ++ni) acc[mi][ni] = (v8f){};

    for (int kk = 0; kk < 512; kk += 32) {
        if (kk + 32 < 512) {
            __builtin_prefetch(Ab + (size_t)lane * 512 + kk + 32, 0, 1);
            __builtin_prefetch(Ab + (size_t)(lane + 32) * 512 + kk + 32, 0, 1);
        }
        v16bf a0 = loadA32(Ab + kk,            512);
        v16bf a1 = loadA32(Ab + 16 * 512 + kk, 512);
        v16bf a2 = loadA32(Ab + 32 * 512 + kk, 512);
        v16bf a3 = loadA32(Ab + 48 * 512 + kk, 512);
        v16bf b0 = loadB32(Wb + kk,            512);
        v16bf b1 = loadB32(Wb + 16 * 512 + kk, 512);
        v16bf b2 = loadB32(Wb + 32 * 512 + kk, 512);
        v16bf b3 = loadB32(Wb + 48 * 512 + kk, 512);
        acc[0][0] = WMMA_BF16(a0, b0, acc[0][0]);
        acc[1][0] = WMMA_BF16(a1, b0, acc[1][0]);
        acc[2][0] = WMMA_BF16(a2, b0, acc[2][0]);
        acc[3][0] = WMMA_BF16(a3, b0, acc[3][0]);
        acc[0][1] = WMMA_BF16(a0, b1, acc[0][1]);
        acc[1][1] = WMMA_BF16(a1, b1, acc[1][1]);
        acc[2][1] = WMMA_BF16(a2, b1, acc[2][1]);
        acc[3][1] = WMMA_BF16(a3, b1, acc[3][1]);
        acc[0][2] = WMMA_BF16(a0, b2, acc[0][2]);
        acc[1][2] = WMMA_BF16(a1, b2, acc[1][2]);
        acc[2][2] = WMMA_BF16(a2, b2, acc[2][2]);
        acc[3][2] = WMMA_BF16(a3, b2, acc[3][2]);
        acc[0][3] = WMMA_BF16(a0, b3, acc[0][3]);
        acc[1][3] = WMMA_BF16(a1, b3, acc[1][3]);
        acc[2][3] = WMMA_BF16(a2, b3, acc[2][3]);
        acc[3][3] = WMMA_BF16(a3, b3, acc[3][3]);
    }

    #pragma unroll
    for (int nt = 0; nt < 4; ++nt) {
        int n = ng * 64 + nt * 16 + nl;
        float bval = bias[n];
        #pragma unroll
        for (int mi = 0; mi < 4; ++mi) {
            #pragma unroll
            for (int e = 0; e < 8; ++e) {
                int m = mt64 * 64 + mi * 16 + e + half * 8;
                int win = m >> 6, tok = m & 63;
                int b = win >> 6, wi = win & 63;
                int wy = wi >> 3, wx = wi & 7, ty = tok >> 3, tx = tok & 7;
                size_t dst = ((size_t)b * 4096 + (size_t)((wy * 8 + ty) * 64 + wx * 8 + tx)) * 512 + n;
                out[dst] = acc[mi][nt][e] + bval;
            }
        }
    }
}

// ---------------- launch ----------------
extern "C" void kernel_launch(void* const* d_in, const int* in_sizes, int n_in,
                              void* d_out, int out_size, void* d_ws, size_t ws_size,
                              hipStream_t stream) {
    const float* x      = (const float*)d_in[0];
    const float* qkv_w  = (const float*)d_in[1];
    const float* qkv_b  = (const float*)d_in[2];
    const float* proj_w = (const float*)d_in[3];
    const float* proj_b = (const float*)d_in[4];
    const float* amask  = (const float*)d_in[5];

    const size_t XWIN = (size_t)65536 * 512;   // bf16 elems
    const size_t QKVW = (size_t)1536 * 512;
    const size_t PRJW = (size_t)512 * 512;
    const size_t QSZ  = (size_t)16384 * 64 * 32;

    ush* xw  = (ush*)d_ws;
    ush* qw  = xw + XWIN;
    ush* pw  = qw + QKVW;
    ush* qbf = pw + PRJW;
    ush* kbf = qbf + QSZ;
    ush* vbf = kbf + QSZ;
    ush* ao  = vbf + QSZ;

    // 1) partition + convert x (8 elems/thread)
    win_part_bf16<<<(unsigned)(XWIN / 8 / 256), 256, 0, stream>>>(x, xw);
    // 2) convert weights
    cvt_bf16<<<(unsigned)((QKVW / 8 + 255) / 256), 256, 0, stream>>>(qkv_w, qw, (int)(QKVW / 8));
    cvt_bf16<<<(unsigned)((PRJW / 8 + 255) / 256), 256, 0, stream>>>(proj_w, pw, (int)(PRJW / 8));
    // 3) QKV GEMM: 1024 M-tiles x 24 N-groups = 24576 waves = 3072 blocks
    qkv_gemm<<<3072, 256, 0, stream>>>(xw, qw, qkv_b, qbf, kbf, vbf);
    // 4) attention: 16384 window-heads, 8 waves/block
    attn_kernel<<<2048, 256, 0, stream>>>(qbf, kbf, vbf, amask, ao);
    // 5) proj GEMM: 1024 M-tiles x 8 N-groups = 8192 waves = 1024 blocks
    proj_gemm<<<1024, 256, 0, stream>>>(ao, pw, proj_b, (float*)d_out);
}
